// GCN3_3066606649549
// MI455X (gfx1250) — compile-verified, matrix operand order checked by reading, not run on hardware
//
#include <hip/hip_runtime.h>
#include <hip/hip_bf16.h>
#include <math.h>

typedef __attribute__((ext_vector_type(16))) __bf16 v16bf;
typedef __attribute__((ext_vector_type(8)))  __bf16 v8bf;
typedef __attribute__((ext_vector_type(8)))  float  v8f;

#define NHID 128
#define NFEAT 602
#define NCLASS 41

// -------------------------------------------------------------------------
// Weight pre-pass: W[K x M] f32 (row-major) -> Bt[Npad x Kpad] bf16,
// Bt[n*Kpad + k] = bf16(W[k*M + n]), zero-padded in both n and k.
// Zero K-padding lets the GEMM skip all K-tail masking (0 * garbage = 0).
// -------------------------------------------------------------------------
__global__ __launch_bounds__(256)
void convert_weight(const float* __restrict__ W, __bf16* __restrict__ Bt,
                    int K, int M, int Kpad, int Npad)
{
    int idx = blockIdx.x * 256 + threadIdx.x;
    int total = Npad * Kpad;
    if (idx >= total) return;
    int n = idx / Kpad;
    int k = idx - n * Kpad;
    float v = (n < M && k < K) ? W[(size_t)k * M + n] : 0.0f;
    Bt[idx] = (__bf16)v;
}

// -------------------------------------------------------------------------
// Dense GEMM  C[N x M] = A[N x K] (f32) * W (via Bt bf16) (+ bias), WMMA.
// 256 threads = 8 waves; each wave computes a 16-row strip over all M cols.
// Workgroup covers 128 rows. NT = number of 16-wide column tiles.
// Double-buffered LDS pipeline: next chunk's global loads are issued into
// registers before the WMMA chain so they overlap compute; one barrier per
// K-chunk. Row index is clamped (masked at store); K-tail handled by Bt's
// zero padding.
// -------------------------------------------------------------------------
template <int NT>
__global__ __launch_bounds__(256)
void gemm_wmma(const float* __restrict__ A, const __bf16* __restrict__ Bt,
               const float* __restrict__ bias, float* __restrict__ C,
               int Nrows, int K, int Kpad, int M)
{
    __shared__ __bf16 lA[2][128 * 32];        // 2 x 8 KB
    __shared__ __bf16 lB[2][NT * 16 * 32];    // 2 x up to 8 KB

    const int tid     = threadIdx.x;
    const int wave    = tid >> 5;
    const int lane    = tid & 31;
    const int rowBase = blockIdx.x * 128;
    const int halfHi  = (lane >> 4) << 3;           // 0 or 8 (K-half select)
    const int mLocal  = (wave << 4) + (lane & 15);  // LDS row of A fragment

    // staging assignment: thread covers 16 contiguous k in one row/col
    const int sRow = tid >> 1;                      // 0..127 (A row / B col)
    const int sKh  = (tid & 1) << 4;                // 0 or 16
    const int gr   = min(rowBase + sRow, Nrows - 1);
    const float*  Arow  = A + (size_t)gr * K;
    const __bf16* BtRow = Bt + (size_t)sRow * Kpad;
    const bool vecA = ((K & 3) == 0);               // float4-alignable rows
    const bool hasB = (sRow < NT * 16);

    float fA[16];
    uint4 fB0, fB1;

    auto loadRegs = [&](int kc) {
        if (vecA && (kc + sKh + 16 <= K)) {
            const float4* p = (const float4*)(Arow + kc + sKh);
            *(float4*)&fA[0]  = p[0];
            *(float4*)&fA[4]  = p[1];
            *(float4*)&fA[8]  = p[2];
            *(float4*)&fA[12] = p[3];
        } else {
            #pragma unroll
            for (int i = 0; i < 16; ++i)
                fA[i] = Arow[min(kc + sKh + i, K - 1)];  // tail hits zero B
        }
        if (hasB) {
            const uint4* bp = (const uint4*)(BtRow + kc + sKh);
            fB0 = bp[0];
            fB1 = bp[1];
        }
        // prime L2 one chunk further ahead (gfx1250 global_prefetch_b8)
        if (kc + 32 < K)
            __builtin_prefetch(Arow + kc + 32 + sKh, 0, 0);
    };

    auto storeTile = [&](int buf) {
        v8bf h0, h1;
        #pragma unroll
        for (int i = 0; i < 8; ++i) {
            h0[i] = (__bf16)fA[i];
            h1[i] = (__bf16)fA[i + 8];
        }
        *(v8bf*)&lA[buf][sRow * 32 + sKh]     = h0;
        *(v8bf*)&lA[buf][sRow * 32 + sKh + 8] = h1;
        if (hasB) {
            *(uint4*)&lB[buf][sRow * 32 + sKh]     = fB0;
            *(uint4*)&lB[buf][sRow * 32 + sKh + 8] = fB1;
        }
    };

    v8f acc[NT] = {};

    loadRegs(0);
    int cur = 0;
    for (int kc = 0; kc < Kpad; kc += 32) {
        storeTile(cur);
        __syncthreads();           // publishes buf[cur]; closes previous compute

        if (kc + 32 < Kpad)
            loadRegs(kc + 32);     // global loads fly during the WMMA chain

        // ---- A fragment: lane holds row (lane&15), K-half by lane>>4 -----
        v8bf alo = *(const v8bf*)&lA[cur][mLocal * 32 + halfHi];
        v8bf ahi = *(const v8bf*)&lA[cur][mLocal * 32 + 16 + halfHi];
        v16bf a  = __builtin_shufflevector(alo, ahi,
                       0,1,2,3,4,5,6,7,8,9,10,11,12,13,14,15);

        #pragma unroll
        for (int t = 0; t < NT; ++t) {
            int n = (t << 4) + (lane & 15);
            v8bf blo = *(const v8bf*)&lB[cur][n * 32 + halfHi];
            v8bf bhi = *(const v8bf*)&lB[cur][n * 32 + 16 + halfHi];
            v16bf b  = __builtin_shufflevector(blo, bhi,
                           0,1,2,3,4,5,6,7,8,9,10,11,12,13,14,15);
            acc[t] = __builtin_amdgcn_wmma_f32_16x16x32_bf16(
                         false, a, false, b, (short)0, acc[t], false, false);
        }
        cur ^= 1;                  // next stores go to the other buffer
    }

    // ---- store: acc[t][r] is (m = r + 8*(lane>>4), n = t*16 + lane&15) ---
    #pragma unroll
    for (int t = 0; t < NT; ++t) {
        int n = (t << 4) + (lane & 15);
        if (n < M) {
            float bv = bias ? bias[n] : 0.0f;
            #pragma unroll
            for (int r = 0; r < 8; ++r) {
                int m = rowBase + (wave << 4) + r + ((lane >> 4) << 3);
                if (m < Nrows) C[(size_t)m * M + n] = acc[t][r] + bv;
            }
        }
    }
}

// -------------------------------------------------------------------------
// SpMM: out[rows[e]] += vals[e] * Hin[cols[e]]   (feature dim = 128)
// One wave per edge; lane handles 4 consecutive features (float4 gather,
// 4x global_atomic_add_f32 scatter). h matrices live in L2 (51 MB < 192 MB).
// -------------------------------------------------------------------------
__global__ __launch_bounds__(256)
void spmm_edges(const int* __restrict__ rows, const int* __restrict__ cols,
                const float* __restrict__ vals, const float* __restrict__ Hin,
                float* __restrict__ Hout, int E)
{
    int e = blockIdx.x * 8 + (threadIdx.x >> 5);
    if (e >= E) return;
    int   lane = threadIdx.x & 31;
    int   r = rows[e];
    int   c = cols[e];
    float v = vals[e];
    float4 h = ((const float4*)(Hin + (size_t)c * NHID))[lane];
    float* o = Hout + (size_t)r * NHID + lane * 4;
    unsafeAtomicAdd(o + 0, v * h.x);
    unsafeAtomicAdd(o + 1, v * h.y);
    unsafeAtomicAdd(o + 2, v * h.z);
    unsafeAtomicAdd(o + 3, v * h.w);
}

// -------------------------------------------------------------------------
// In-place bias + LayerNorm + ReLU over 128 features; one wave32 per row.
// -------------------------------------------------------------------------
__global__ __launch_bounds__(256)
void bias_ln_relu(float* __restrict__ H, const float* __restrict__ bias,
                  const float* __restrict__ g, const float* __restrict__ beta,
                  int Nrows)
{
    int row = blockIdx.x * 8 + (threadIdx.x >> 5);
    if (row >= Nrows) return;
    int lane = threadIdx.x & 31;
    float4* hp = (float4*)(H + (size_t)row * NHID);
    float4 x = hp[lane];
    float4 bv = ((const float4*)bias)[lane];
    x.x += bv.x; x.y += bv.y; x.z += bv.z; x.w += bv.w;
    float s  = x.x + x.y + x.z + x.w;
    float sq = x.x * x.x + x.y * x.y + x.z * x.z + x.w * x.w;
    #pragma unroll
    for (int m = 16; m >= 1; m >>= 1) {
        s  += __shfl_xor(s,  m, 32);
        sq += __shfl_xor(sq, m, 32);
    }
    float mu  = s * (1.0f / 128.0f);
    float var = sq * (1.0f / 128.0f) - mu * mu;
    float rs  = rsqrtf(var + 1e-5f);
    float4 gv = ((const float4*)g)[lane];
    float4 bb = ((const float4*)beta)[lane];
    float4 y;
    y.x = fmaxf(0.0f, (x.x - mu) * rs * gv.x + bb.x);
    y.y = fmaxf(0.0f, (x.y - mu) * rs * gv.y + bb.y);
    y.z = fmaxf(0.0f, (x.z - mu) * rs * gv.z + bb.z);
    y.w = fmaxf(0.0f, (x.w - mu) * rs * gv.w + bb.w);
    hp[lane] = y;
}

// -------------------------------------------------------------------------
// In-place log_softmax over 41 classes; one wave32 per row (lane covers
// columns {lane, lane+32}).
// -------------------------------------------------------------------------
__global__ __launch_bounds__(256)
void log_softmax41(float* __restrict__ O, int Nrows)
{
    int row = blockIdx.x * 8 + (threadIdx.x >> 5);
    if (row >= Nrows) return;
    int lane = threadIdx.x & 31;
    float* p = O + (size_t)row * NCLASS;
    float v0 = (lane < NCLASS)        ? p[lane]      : -3.0e38f;
    float v1 = ((lane + 32) < NCLASS) ? p[lane + 32] : -3.0e38f;
    float mx = fmaxf(v0, v1);
    #pragma unroll
    for (int m = 16; m >= 1; m >>= 1) mx = fmaxf(mx, __shfl_xor(mx, m, 32));
    float s = ((lane < NCLASS) ? expf(v0 - mx) : 0.0f) +
              (((lane + 32) < NCLASS) ? expf(v1 - mx) : 0.0f);
    #pragma unroll
    for (int m = 16; m >= 1; m >>= 1) s += __shfl_xor(s, m, 32);
    float lse = mx + logf(s);
    if (lane < NCLASS)        p[lane]      = v0 - lse;
    if ((lane + 32) < NCLASS) p[lane + 32] = v1 - lse;
}

__global__ void fill_zero(float* __restrict__ p, size_t n)
{
    size_t i      = (size_t)blockIdx.x * blockDim.x + threadIdx.x;
    size_t stride = (size_t)gridDim.x * blockDim.x;
    for (; i < n; i += stride) p[i] = 0.0f;
}

// -------------------------------------------------------------------------
extern "C" void kernel_launch(void* const* d_in, const int* in_sizes, int n_in,
                              void* d_out, int out_size, void* d_ws, size_t ws_size,
                              hipStream_t stream)
{
    (void)n_in; (void)out_size; (void)ws_size;
    const float* x    = (const float*)d_in[0];
    const int*   rows = (const int*)  d_in[1];
    const int*   cols = (const int*)  d_in[2];
    const float* vals = (const float*)d_in[3];
    const float* W1   = (const float*)d_in[4];
    const float* b1   = (const float*)d_in[5];
    const float* W2   = (const float*)d_in[6];
    const float* b2   = (const float*)d_in[7];
    const float* W3   = (const float*)d_in[8];
    const float* b3   = (const float*)d_in[9];
    const float* ln_g = (const float*)d_in[10];
    const float* ln_b = (const float*)d_in[11];
    const float* Wout = (const float*)d_in[12];
    const float* bout = (const float*)d_in[13];

    const int N = in_sizes[0] / NFEAT;
    const int E = in_sizes[1] / 3;

    const int KP1 = ((NFEAT + 31) / 32) * 32;   // 608
    const int KP2 = NHID;                       // 128

    float*  bufA = (float*)d_ws;                   // N x 128 (dense GEMM out)
    float*  bufB = bufA + (size_t)N * NHID;        // N x 128 (spmm accum / h)
    __bf16* Bt1  = (__bf16*)(bufB + (size_t)N * NHID);  // 128 x 608
    __bf16* Bt2  = Bt1 + 128 * KP1;                     // 128 x 128
    __bf16* Bt3  = Bt2 + 128 * KP2;                     // 128 x 128
    __bf16* BtO  = Bt3 + 128 * KP2;                     // 48  x 128

    dim3 blk(256);
    dim3 gemmGrid((N + 127) / 128);
    dim3 rowGrid((N + 7) / 8);
    dim3 edgeGrid((E + 7) / 8);
    const size_t hElems = (size_t)N * NHID;

    // ---- weight pre-conversion (tiny) ----
    convert_weight<<<(128 * KP1 + 255) / 256, blk, 0, stream>>>(W1, Bt1, NFEAT, NHID, KP1, 128);
    convert_weight<<<(128 * KP2 + 255) / 256, blk, 0, stream>>>(W2, Bt2, NHID, NHID, KP2, 128);
    convert_weight<<<(128 * KP2 + 255) / 256, blk, 0, stream>>>(W3, Bt3, NHID, NHID, KP2, 128);
    convert_weight<<<(48  * KP2 + 255) / 256, blk, 0, stream>>>(Wout, BtO, NHID, NCLASS, KP2, 48);

    const int*    lrows[3] = { rows, rows + E, rows + 2 * E };
    const int*    lcols[3] = { cols, cols + E, cols + 2 * E };
    const float*  lvals[3] = { vals, vals + E, vals + 2 * E };
    const __bf16* lBt[3]   = { Bt1, Bt2, Bt3 };
    const float*  lb[3]    = { b1, b2, b3 };
    const int     lK[3]    = { NFEAT, NHID, NHID };
    const int     lKP[3]   = { KP1, KP2, KP2 };
    const float*  src = x;

    for (int layer = 0; layer < 3; ++layer) {
        gemm_wmma<8><<<gemmGrid, blk, 0, stream>>>(src, lBt[layer], nullptr, bufA,
                                                   N, lK[layer], lKP[layer], NHID);
        fill_zero<<<2048, blk, 0, stream>>>(bufB, hElems);
        spmm_edges<<<edgeGrid, blk, 0, stream>>>(lrows[layer], lcols[layer],
                                                 lvals[layer], bufA, bufB, E);
        bias_ln_relu<<<rowGrid, blk, 0, stream>>>(bufB, lb[layer], ln_g, ln_b, N);
        src = bufB;
    }

    gemm_wmma<3><<<gemmGrid, blk, 0, stream>>>(bufB, BtO, bout, (float*)d_out,
                                               N, NHID, KP2, NCLASS);
    log_softmax41<<<rowGrid, blk, 0, stream>>>((float*)d_out, N);
}
